// FragmentAwareGCNEncoder_28037546508389
// MI455X (gfx1250) — matrix operand order repaired and checked by reading, not compiled, not run on hardware
//
#include <hip/hip_runtime.h>

#define N_NODES 10240
#define B_GR    64
#define NPG     160
#define E_EDGES 320000
#define K_FRAG  16
#define DIN     25
#define D       256
#define L_LAYERS 4

typedef __attribute__((ext_vector_type(16))) __bf16          v16bf;
typedef __attribute__((ext_vector_type(16))) unsigned short  v16u;
typedef __attribute__((ext_vector_type(8)))  float           v8f;

__device__ __forceinline__ unsigned short f2bf(float f) {
  union { float f; unsigned u; } v; v.f = f;
  unsigned r = v.u + 0x7FFFu + ((v.u >> 16) & 1u);   // round-to-nearest-even
  return (unsigned short)(r >> 16);
}
__device__ __forceinline__ float bf2f(unsigned short h) {
  union { unsigned u; float f; } v; v.u = ((unsigned)h) << 16;
  return v.f;
}
// load 8 contiguous bf16 from p0 -> elems 0..7, 8 from p1 -> elems 8..15
__device__ __forceinline__ v16bf load_bf16x16(const unsigned short* p0,
                                              const unsigned short* p1) {
  union { v16u v; uint4 q[2]; } u;
  u.q[0] = *(const uint4*)p0;
  u.q[1] = *(const uint4*)p1;
  return __builtin_bit_cast(v16bf, u.v);
}

// ---------------- utility kernels ----------------
__global__ void fa_fill_zero(float* p, int n) {
  int i = blockIdx.x * blockDim.x + threadIdx.x;
  if (i < n) p[i] = 0.f;
}
__global__ void fa_init_scalar(float* p) { *p = 0.f; }

__global__ void fa_degree(const int* __restrict__ ei, float* __restrict__ deg) {
  int e = blockIdx.x * blockDim.x + threadIdx.x;
  if (e < E_EDGES) atomicAdd(&deg[ei[E_EDGES + e]], 1.0f);
}
__global__ void fa_dinv(const float* __restrict__ deg, float* __restrict__ dinv) {
  int n = blockIdx.x * blockDim.x + threadIdx.x;
  if (n < N_NODES) dinv[n] = rsqrtf(fmaxf(deg[n], 1.0f));
}
__global__ void fa_norm(const int* __restrict__ ei, const float* __restrict__ dinv,
                        float* __restrict__ norm) {
  int e = blockIdx.x * blockDim.x + threadIdx.x;
  if (e < E_EDGES) norm[e] = dinv[ei[e]] * dinv[ei[E_EDGES + e]];
}

// convert f32 [3,256,256] row-major W[l][k][d] -> bf16 transposed Wt[l][d][k]
__global__ void fa_prep_weights(const float* __restrict__ Wi, const float* __restrict__ Wo,
                                unsigned short* __restrict__ WtI, unsigned short* __restrict__ WtO) {
  int idx = blockIdx.x * blockDim.x + threadIdx.x;   // 3*65536
  if (idx >= 3 * D * D) return;
  int l = idx >> 16, r = idx & 0xFFFF;
  int k = r >> 8, d = r & 255;
  WtI[l * D * D + d * D + k] = f2bf(Wi[l * D * D + k * D + d]);
  WtO[l * D * D + d * D + k] = f2bf(Wo[l * D * D + k * D + d]);
}

// ---------------- layer 0: [10240,25] @ [25,256] x2 (VALU; K too skinny for WMMA) ----
__global__ void fa_layer0(const float* __restrict__ x,
                          const float* __restrict__ Wi, const float* __restrict__ Wo,
                          unsigned short* __restrict__ hi, unsigned short* __restrict__ ho) {
  int n = blockIdx.x, d = threadIdx.x;
  __shared__ float xs[DIN];
  if (d < DIN) xs[d] = x[n * DIN + d];
  __syncthreads();
  float si = 0.f, so = 0.f;
#pragma unroll
  for (int k = 0; k < DIN; k++) {
    si += xs[k] * Wi[k * D + d];
    so += xs[k] * Wo[k * D + d];
  }
  hi[n * D + d] = f2bf(si);
  ho[n * D + d] = f2bf(so);
}

// ---------------- layers 1..3: dual GEMM via v_wmma_f32_16x16x32_bf16 ----------------
// One wave per 16x16 output tile; A fragment shared by Wi and Wo WMMAs.
__global__ __launch_bounds__(256) void fa_gemm_dual_wmma(
    const unsigned short* __restrict__ hb,   // [N,256] bf16 row-major
    const unsigned short* __restrict__ WiT,  // [256,256] bf16, WiT[col*256+k]
    const unsigned short* __restrict__ WoT,
    unsigned short* __restrict__ hi, unsigned short* __restrict__ ho) {
  int wave = (blockIdx.x * blockDim.x + threadIdx.x) >> 5;  // 640*16 waves
  int lane = threadIdx.x & 31;
  int tn = wave & 15, tm = wave >> 4;
  int l15 = lane & 15, h16 = lane >> 4;
  const unsigned short* arow = hb  + (tm * 16 + l15) * D + h16 * 16;  // A: M=l15
  const unsigned short* bci  = WiT + (tn * 16 + l15) * D + h16 * 16;  // B: N=l15
  const unsigned short* bco  = WoT + (tn * 16 + l15) * D + h16 * 16;
  v8f ci = {}, co = {};
#pragma unroll
  for (int kk = 0; kk < D; kk += 32) {
    v16bf a  = load_bf16x16(arow + kk, arow + kk + 8);    // K contiguous 16
    v16bf bi = load_bf16x16(bci + kk, bci + kk + 8);      // K contiguous 16
    v16bf bo = load_bf16x16(bco + kk, bco + kk + 8);
    ci = __builtin_amdgcn_wmma_f32_16x16x32_bf16(false, a, false, bi, (short)0, ci, false, false);
    co = __builtin_amdgcn_wmma_f32_16x16x32_bf16(false, a, false, bo, (short)0, co, false, false);
  }
  int orow = tm * 16 + h16 * 8, col = tn * 16 + l15;      // C: VGPR r -> M=r(+8)
#pragma unroll
  for (int r = 0; r < 8; r++) {
    hi[(orow + r) * D + col] = f2bf(ci[r]);
    ho[(orow + r) * D + col] = f2bf(co[r]);
  }
}

// ---------------- edge gather * norm -> atomic scatter (L2-resident) ----------------
__global__ void fa_edge_scatter(const unsigned short* __restrict__ hi,
                                const unsigned short* __restrict__ ho,
                                const int* __restrict__ ei,
                                const unsigned char* __restrict__ mask,
                                const float* __restrict__ norm,
                                float* __restrict__ agg) {
  int idx = blockIdx.x * blockDim.x + threadIdx.x;        // E * 64 threads
  if (idx >= E_EDGES * 64) return;
  int e = idx >> 6, c = (idx & 63) << 2;
  int s = ei[e], d = ei[E_EDGES + e];
  const unsigned short* p = (mask[e] ? hi : ho) + s * D + c;
  uint2 w = *(const uint2*)p;                             // 4 bf16
  float nm = norm[e];
  float* a = agg + d * D + c;
  atomicAdd(a + 0, bf2f((unsigned short)(w.x & 0xFFFF)) * nm);
  atomicAdd(a + 1, bf2f((unsigned short)(w.x >> 16)) * nm);
  atomicAdd(a + 2, bf2f((unsigned short)(w.y & 0xFFFF)) * nm);
  atomicAdd(a + 3, bf2f((unsigned short)(w.y >> 16)) * nm);
}

// ---------------- bias + layernorm + relu -> bf16 activations ----------------
__global__ void fa_post_layer(const float* __restrict__ agg,
                              const float* __restrict__ bi, const float* __restrict__ bo,
                              const float* __restrict__ g, const float* __restrict__ b,
                              unsigned short* __restrict__ hb) {
  int n = blockIdx.x, d = threadIdx.x;
  float v = agg[n * D + d] + bi[d] + bo[d];
  __shared__ float rs[D], rq[D];
  rs[d] = v; rq[d] = v * v;
  __syncthreads();
  for (int s = 128; s > 0; s >>= 1) {
    if (d < s) { rs[d] += rs[d + s]; rq[d] += rq[d + s]; }
    __syncthreads();
  }
  float mean = rs[0] * (1.0f / D);
  float var  = rq[0] * (1.0f / D) - mean * mean;
  float y = (v - mean) * rsqrtf(var + 1e-5f) * g[d] + b[d];
  hb[n * D + d] = f2bf(fmaxf(y, 0.f));
}

// ---------------- fragment pooling: per graph s^T(16x160) @ h(160x256) via WMMA ------
__global__ __launch_bounds__(256) void fa_frag_wmma(const float* __restrict__ s,
                                                    const unsigned short* __restrict__ hb,
                                                    float* __restrict__ frag_raw) {
  int wave = (blockIdx.x * blockDim.x + threadIdx.x) >> 5;  // 64*16 waves
  int lane = threadIdx.x & 31;
  int g = wave >> 4, tn = wave & 15;
  int l15 = lane & 15, h16 = lane >> 4;
  int col = tn * 16 + l15;
  v8f c = {};
  for (int kk = 0; kk < NPG; kk += 32) {
    v16u au;
    int nA = g * NPG + kk + h16 * 16;
#pragma unroll
    for (int i = 0; i < 8; i++) {
      au[i]     = f2bf(s[(nA + i) * K_FRAG + l15]);        // A[M=frag l15][K=node]
      au[8 + i] = f2bf(s[(nA + 8 + i) * K_FRAG + l15]);
    }
    v16bf a = __builtin_bit_cast(v16bf, au);
    const unsigned short* bp = hb + (g * NPG + kk + h16 * 16) * D + col;
    v16u bu;
#pragma unroll
    for (int i = 0; i < 16; i++) bu[i] = bp[i * D];        // B[K=node][N=col]
    v16bf b = __builtin_bit_cast(v16bf, bu);
    c = __builtin_amdgcn_wmma_f32_16x16x32_bf16(false, a, false, b, (short)0, c, false, false);
  }
  int orow = g * K_FRAG + h16 * 8;
#pragma unroll
  for (int r = 0; r < 8; r++) frag_raw[(orow + r) * D + col] = c[r];
}

__global__ void fa_frag_ln(const float* __restrict__ fr,
                           const float* __restrict__ g, const float* __restrict__ b,
                           float* __restrict__ out) {
  int n = blockIdx.x, d = threadIdx.x;                    // n over B*K
  float v = fr[n * D + d];
  __shared__ float rs[D], rq[D];
  rs[d] = v; rq[d] = v * v;
  __syncthreads();
  for (int s = 128; s > 0; s >>= 1) {
    if (d < s) { rs[d] += rs[d + s]; rq[d] += rq[d + s]; }
    __syncthreads();
  }
  float mean = rs[0] * (1.0f / D);
  float var  = rq[0] * (1.0f / D) - mean * mean;
  out[n * D + d] = (v - mean) * rsqrtf(var + 1e-5f) * g[d] + b[d];
}

__global__ void fa_frag_mask(const float* __restrict__ s, float* __restrict__ out) {
  int idx = blockIdx.x * blockDim.x + threadIdx.x;        // B*K
  if (idx >= B_GR * K_FRAG) return;
  int g = idx >> 4, k = idx & 15;
  float sum = 0.f;
  for (int n = 0; n < NPG; n++) sum += s[(g * NPG + n) * K_FRAG + k];
  out[idx] = (sum > 0.f) ? 1.f : 0.f;
}

__global__ void fa_abs_sum(const float* __restrict__ p, int n, float* __restrict__ acc) {
  __shared__ float red[256];
  int tid = threadIdx.x;
  float s = 0.f;
  for (int i = blockIdx.x * blockDim.x + tid; i < n; i += gridDim.x * blockDim.x)
    s += fabsf(p[i]);
  red[tid] = s; __syncthreads();
  for (int w = 128; w > 0; w >>= 1) {
    if (tid < w) red[tid] += red[tid + w];
    __syncthreads();
  }
  if (tid == 0) atomicAdd(acc, red[0]);
}

// ------------------------------- launch -------------------------------
extern "C" void kernel_launch(void* const* d_in, const int* in_sizes, int n_in,
                              void* d_out, int out_size, void* d_ws, size_t ws_size,
                              hipStream_t stream) {
  (void)in_sizes; (void)n_in; (void)out_size; (void)ws_size;
  const float*         x      = (const float*)d_in[0];
  const int*           ei     = (const int*)d_in[1];
  const unsigned char* mask   = (const unsigned char*)d_in[2];   // numpy bool = 1 byte
  const float*         s      = (const float*)d_in[3];
  const float*         W_in0  = (const float*)d_in[5];
  const float*         W_out0 = (const float*)d_in[6];
  const float*         b_in0  = (const float*)d_in[7];
  const float*         b_out0 = (const float*)d_in[8];
  const float*         W_in   = (const float*)d_in[9];
  const float*         W_out  = (const float*)d_in[10];
  const float*         b_in   = (const float*)d_in[11];
  const float*         b_out  = (const float*)d_in[12];
  const float*         ln_g   = (const float*)d_in[13];
  const float*         ln_b   = (const float*)d_in[14];
  const float*         fbn_g  = (const float*)d_in[15];
  const float*         fbn_b  = (const float*)d_in[16];
  float* out = (float*)d_out;

  char* ws = (char*)d_ws;
  size_t off = 0;
  auto alloc = [&](size_t bytes) { char* p = ws + off; off = (off + bytes + 255) & ~(size_t)255; return p; };
  float*          deg  = (float*)alloc(N_NODES * 4);
  float*          dinv = (float*)alloc(N_NODES * 4);
  float*          norm = (float*)alloc((size_t)E_EDGES * 4);
  unsigned short* hb   = (unsigned short*)alloc((size_t)N_NODES * D * 2);
  unsigned short* hi   = (unsigned short*)alloc((size_t)N_NODES * D * 2);
  unsigned short* ho   = (unsigned short*)alloc((size_t)N_NODES * D * 2);
  float*          agg  = (float*)alloc((size_t)N_NODES * D * 4);
  unsigned short* WtI  = (unsigned short*)alloc((size_t)3 * D * D * 2);
  unsigned short* WtO  = (unsigned short*)alloc((size_t)3 * D * D * 2);
  float*          fraw = (float*)alloc((size_t)B_GR * K_FRAG * D * 4);

  // degree -> dinv -> edge norm
  fa_fill_zero<<<(N_NODES + 255) / 256, 256, 0, stream>>>(deg, N_NODES);
  fa_degree<<<E_EDGES / 256, 256, 0, stream>>>(ei, deg);
  fa_dinv<<<N_NODES / 256, 256, 0, stream>>>(deg, dinv);
  fa_norm<<<E_EDGES / 256, 256, 0, stream>>>(ei, dinv, norm);

  // bf16 transposed weights for WMMA B-operands
  fa_prep_weights<<<(3 * D * D) / 256, 256, 0, stream>>>(W_in, W_out, WtI, WtO);

  for (int l = 0; l < L_LAYERS; l++) {
    if (l == 0) {
      fa_layer0<<<N_NODES, 256, 0, stream>>>(x, W_in0, W_out0, hi, ho);
    } else {
      fa_gemm_dual_wmma<<<(N_NODES / 16) * (D / 16) / 8, 256, 0, stream>>>(
          hb, WtI + (size_t)(l - 1) * D * D, WtO + (size_t)(l - 1) * D * D, hi, ho);
    }
    fa_fill_zero<<<(N_NODES * D) / 256, 256, 0, stream>>>(agg, N_NODES * D);
    fa_edge_scatter<<<(E_EDGES * 64) / 256, 256, 0, stream>>>(hi, ho, ei, mask, norm, agg);
    const float* bi = (l == 0) ? b_in0  : b_in  + (size_t)(l - 1) * D;
    const float* bo = (l == 0) ? b_out0 : b_out + (size_t)(l - 1) * D;
    fa_post_layer<<<N_NODES, 256, 0, stream>>>(agg, bi, bo, ln_g + (size_t)l * D, ln_b + (size_t)l * D, hb);
  }

  // fragment pooling (WMMA), layernorm, mask, reg loss
  fa_frag_wmma<<<(B_GR * 16) / 8, 256, 0, stream>>>(s, hb, fraw);
  fa_frag_ln<<<B_GR * K_FRAG, 256, 0, stream>>>(fraw, fbn_g, fbn_b, out);
  fa_frag_mask<<<(B_GR * K_FRAG + 255) / 256, 256, 0, stream>>>(s, out + B_GR * K_FRAG * D);

  float* reg = out + B_GR * K_FRAG * D + B_GR * K_FRAG;
  fa_init_scalar<<<1, 1, 0, stream>>>(reg);
  fa_abs_sum<<<26, 256, 0, stream>>>(W_out0, DIN * D, reg);
  fa_abs_sum<<<1, 256, 0, stream>>>(b_out0, D, reg);
  fa_abs_sum<<<256, 256, 0, stream>>>(W_out, 3 * D * D, reg);
  fa_abs_sum<<<3, 256, 0, stream>>>(b_out, 3 * D, reg);
}
